// PeptidePocketConvLayer_11072425689946
// MI455X (gfx1250) — matrix operand order repaired and checked by reading, not compiled, not run on hardware
//
#include <hip/hip_runtime.h>

// CDNA5 / gfx1250 wave32 WMMA types
typedef __attribute__((ext_vector_type(16))) _Float16 v16h;
typedef __attribute__((ext_vector_type(8)))  _Float16 v8h;
typedef __attribute__((ext_vector_type(8)))  float    v8f;
typedef __attribute__((ext_vector_type(4)))  float    v4f;

#define NTHREADS    256      // 8 wave32s
#define ROWS_PER_WG 256      // one (b,p) row per thread; grid divides exactly
#define NCODES      20       // alphabet size == number of distinct filters
#define ALPHA       20       // conv input length (amino acids)
#define FILT        9        // filter taps
#define KOUT        28       // full-conv output length = ALPHA + FILT - 1
#define PPB         34       // pockets per batch
#define PEP_FLOATS  300      // 15 * 20 floats per batch
#define KPAD        32       // K padded 20 -> 32 for 16x16x32 WMMA
#define MAX_SLOTS   560      // 256 rows + 20 buckets * 15 pad, rounded up
#define MAXB        9        // max batches a 256-row window can span
#define MAXTILES    40       // sum ceil(cnt_c/16) <= (256 + 20*15)/16 = 34

// ---- gfx1250 async global->LDS copy (ASYNCcnt-tracked, GV mode) ----------
__device__ __forceinline__ unsigned lds_addr32(const void* p) {
    // LDS-relative byte offset for the async-op VDST operand
    return (unsigned)(unsigned long long)
        (__attribute__((address_space(3))) const void*)p;
}
__device__ __forceinline__ void async_g2l_b128(unsigned lds_off, const void* g) {
    asm volatile("global_load_async_to_lds_b128 %0, %1, off"
                 :: "v"(lds_off), "v"((unsigned long long)(size_t)g) : "memory");
}
__device__ __forceinline__ void wait_asynccnt0() {
    asm volatile("s_wait_asynccnt 0x0" ::: "memory");
}

__global__ void __launch_bounds__(NTHREADS)
peptide_pocket_conv_wmma(const float* __restrict__ pep,
                         const int*   __restrict__ pocket,
                         const float* __restrict__ kern,
                         float* __restrict__ out)
{
    // ~50 KB static LDS -> ~6 workgroups per 320KB WGP
    __shared__ _Float16 __align__(16) sAgg[MAX_SLOTS * KPAD];  // bucketed A panel (f16)
    __shared__ float    __align__(16) sPep[MAXB * PEP_FLOATS]; // staged peptide blocks
    __shared__ float    __align__(16) sKern[NCODES * FILT];    // filter bank
    __shared__ int      sPerm[MAX_SLOTS];                      // bucket slot -> local row
    __shared__ int      sCnt[NCODES];
    __shared__ int      sOff[NCODES];
    __shared__ int      sTileCode[MAXTILES];
    __shared__ int      sTileRow[MAXTILES];
    __shared__ int      sNT;

    const int      tid = threadIdx.x;
    const unsigned r0  = (unsigned)blockIdx.x * ROWS_PER_WG;   // first global (b,p) row
    const unsigned b0  = r0 / PPB;
    const unsigned be  = (r0 + ROWS_PER_WG - 1) / PPB;
    const int      nb  = (int)(be - b0) + 1;                   // <= 9 batches

    __builtin_prefetch(pocket + r0, 0, 0);   // global_prefetch_b8, in-bounds

    if (tid < NCODES) sCnt[tid] = 0;
    __syncthreads();

    // ---- phase 1a: per-row code, bucket histogram ----------------------
    const unsigned r    = r0 + tid;
    const int      code = pocket[r];                    // in [0, 20)
    const int      rank = atomicAdd(&sCnt[code], 1);

    // ---- async-stage peptide blocks + filter bank (ASYNCcnt path) ------
    {
        const char* gsrc = (const char*)(pep + (size_t)b0 * PEP_FLOATS);
        const unsigned pOff = lds_addr32(sPep);
        const int n4 = nb * (PEP_FLOATS / 4);            // 16B chunks, <= 675
        for (int i = tid; i < n4; i += NTHREADS)
            async_g2l_b128(pOff + 16u * i, gsrc + 16u * i);
        if (tid < (NCODES * FILT) / 4)                   // 180 floats = 45 x b128
            async_g2l_b128(lds_addr32(sKern) + 16u * tid,
                           (const char*)kern + 16u * tid);
    }
    // zero padded A panel (overlaps with async copies in flight)
    {
        v4f  z   = {};
        v4f* dst = (v4f*)sAgg;
        for (int i = tid; i < (MAX_SLOTS * KPAD) / 8; i += NTHREADS) dst[i] = z;
    }
    wait_asynccnt0();          // drain this wave's ASYNCcnt before the barrier
    __syncthreads();

    // ---- phase 1b: 16-padded bucket offsets + tile table (serial, tiny) -
    if (tid == 0) {
        int acc = 0, t = 0;
        for (int c = 0; c < NCODES; ++c) {
            sOff[c] = acc;
            const int tiles = (sCnt[c] + 15) >> 4;
            for (int k = 0; k < tiles; ++k) { sTileCode[t] = c; sTileRow[t] = acc + k * 16; ++t; }
            acc += tiles * 16;
        }
        sNT = t;
    }
    __syncthreads();

    // ---- phase 1c: aggregate mask rows, scatter f16 row into its bucket -
    {
        const unsigned b  = r / PPB;
        const int      p  = (int)(r - b * PPB);
        const int      j1 = p % 9;
        const int      j2 = 3 * (p % 3);
        const float* __restrict__ p1 = sPep + (b - b0) * PEP_FLOATS + j1 * ALPHA;
        const float* __restrict__ p2 = sPep + (b - b0) * PEP_FLOATS + j2 * ALPHA;
        const float  m2  = (j1 == j2) ? 0.f : 1.f;
        const int    dst = sOff[code] + rank;
        sPerm[dst] = (int)(r - r0);
        unsigned* arow = (unsigned*)(sAgg + dst * KPAD);   // cols 20..31 stay zero
        #pragma unroll
        for (int a = 0; a < ALPHA; a += 2) {
            union { _Float16 h[2]; unsigned u; } pk;
            pk.h[0] = (_Float16)(p1[a]     + m2 * p2[a]);
            pk.h[1] = (_Float16)(p1[a + 1] + m2 * p2[a + 1]);
            arow[a >> 1] = pk.u;
        }
    }
    __syncthreads();

    // ---- phase 2: per-bucket GEMM tiles, D = A(16x32 f16) x B(32x16 f16) -
    const int wave = tid >> 5;
    const int lane = tid & 31;
    const int col  = lane & 15;
    const int hsel = lane >> 4;
    const int NT   = sNT;                 // wave-uniform -> EXEC all-1 at WMMA

    for (int t = wave; t < NT; t += NTHREADS / 32) {
        const int c       = sTileCode[t];
        const int rowbase = sTileRow[t];

        // A operand: lanes 0-15 hold K 0..15, lanes 16-31 K 16..31
        const _Float16* ap = sAgg + (rowbase + col) * KPAD;
        union { v16h v; v8h h[2]; } A;
        A.h[0] = *(const v8h*)(ap + hsel * 16);
        A.h[1] = *(const v8h*)(ap + hsel * 16 + 8);

        // B operands (Toeplitz of filter c): lane holds K = hsel*16 + j, col = lane&15
        const float* kc = sKern + c * FILT;
        const int    kb = hsel * 16;
        v16h B0, B1;
        #pragma unroll
        for (int j = 0; j < 16; ++j) {
            const int a  = kb + j;
            const int d0 = col - a;
            const int d1 = col + 16 - a;
            float f0 = 0.f, f1 = 0.f;
            if (a < ALPHA) {
                if (d0 >= 0 && d0 < FILT) f0 = kc[d0];
                if (d1 >= 0 && d1 < FILT) f1 = kc[d1];
            }
            B0[j] = (_Float16)f0;
            B1[j] = (_Float16)f1;
        }

        v8f acc0 = {};
        v8f acc1 = {};
        acc0 = __builtin_amdgcn_wmma_f32_16x16x32_f16(false, A.v, false, B0,
                                                      (short)0, acc0, false, false);
        acc1 = __builtin_amdgcn_wmma_f32_16x16x32_f16(false, A.v, false, B1,
                                                      (short)0, acc1, false, false);

        // D layout: VGPR v, lanes0-15 -> row v, lanes16-31 -> row v+8, N = lane&15.
        // Per VGPR, 16 lanes write 16 contiguous floats of one output row (64B line).
        const int nval = sCnt[c] - (rowbase - sOff[c]);   // valid rows in tile
        #pragma unroll
        for (int v = 0; v < 8; ++v) {
            const int m = hsel * 8 + v;
            if (m < nval) {
                const unsigned rr = r0 + (unsigned)sPerm[rowbase + m];
                float* o = out + (size_t)rr * KOUT;
                o[col] = acc0[v];
                if (col < (KOUT - 16)) o[16 + col] = acc1[v];
            }
        }
    }
}

extern "C" void kernel_launch(void* const* d_in, const int* in_sizes, int n_in,
                              void* d_out, int out_size, void* d_ws, size_t ws_size,
                              hipStream_t stream) {
    const float* pep    = (const float*)d_in[0];   // [B,15,20] f32
    const int*   pocket = (const int*)  d_in[1];   // [B,34]    i32
    const float* kern   = (const float*)d_in[2];   // [20,9]    f32
    float*       out    = (float*)d_out;           // [B,34,28] f32

    const int total_rows = in_sizes[1];            // B*34 = 8,912,896 = 34816 * 256
    const int grid = total_rows / ROWS_PER_WG;     // divides exactly for this shape
    peptide_pocket_conv_wmma<<<grid, NTHREADS, 0, stream>>>(pep, pocket, kern, out);
}
